// LoRALinear_17042430231317
// MI455X (gfx1250) — compile-verified, matrix operand order checked by reading, not run on hardware
//
#include <hip/hip_runtime.h>
#include <hip/hip_bf16.h>

// ---------------------------------------------------------------------------
// QLoRA linear for MI455X (gfx1250, wave32, WMMA):
//   out = x @ dequant_nf4(W)^T + (alpha/rank) * (x @ A^T) @ B^T
// Prepack W (NF4 -> bf16) and x (f32 -> bf16) once, fold LoRA into 64 extra
// K-columns (rank 16 + zero pad), one bf16 WMMA GEMM with K = 4160.
// Both bf16 operands (~102 MB) fit in the 192 MB L2 -> tile re-reads on-chip.
// Tiles are staged with CDNA5 async global->LDS DMA (ASYNCcnt) overlapped
// with the v_wmma stream; BK=64 so each barrier pair covers 16 WMMAs.
// ---------------------------------------------------------------------------

#define IN_DIM   4096
#define OUT_DIM  4096
#define RANK     16
#define NTOK     8192
#define KPAD     64
#define KAUG     (IN_DIM + KPAD)   // 4160
#define BM 128
#define BN 128
#define BK 64
#define KTILES   (KAUG / BK)       // 65

typedef __attribute__((ext_vector_type(16))) __bf16 bf16x16;
typedef __attribute__((ext_vector_type(8)))  float  v8f;

__constant__ float NF4_LUT[16] = {
    -1.0f, -0.6961928009986877f, -0.5250730514526367f, -0.39491748809814453f,
    -0.28444138169288635f, -0.18477343022823334f, -0.09105003625154495f, 0.0f,
    0.07958029955625534f, 0.16093020141124725f, 0.24611230194568634f,
    0.33791524171829224f, 0.44070982933044434f, 0.5626170039176941f,
    0.7229568362236023f, 1.0f};

__device__ __forceinline__ unsigned short f32_to_bf16_rte(float f) {
  union { float f; unsigned u; } v; v.f = f;
  unsigned r = v.u + 0x7FFFu + ((v.u >> 16) & 1u);   // round-to-nearest-even
  return (unsigned short)(r >> 16);
}

// --- prep 1: NF4 dequant -> bf16 Waug[OUT_DIM][KAUG]; tail = 2.0 * B -------
__global__ __launch_bounds__(256) void nf4_dequant_w(
    const int* __restrict__ codes, const float* __restrict__ scalers,
    const float* __restrict__ lora_b, unsigned short* __restrict__ Wb) {
  size_t idx = (size_t)blockIdx.x * blockDim.x + threadIdx.x;
  const size_t total = (size_t)OUT_DIM * KAUG;
  if (idx >= total) return;
  int o = (int)(idx / KAUG);
  int k = (int)(idx % KAUG);
  float v;
  if (k < IN_DIM) {
    size_t w = (size_t)o * IN_DIM + (size_t)k;
    v = NF4_LUT[codes[w] & 15] * scalers[w >> 6];    // block = 64 elems of a row
  } else {
    int r = k - IN_DIM;
    v = (r < RANK) ? 2.0f * lora_b[o * RANK + r] : 0.0f;  // alpha/rank = 2.0
  }
  Wb[idx] = f32_to_bf16_rte(v);
}

// --- prep 2: x f32 -> bf16 Xaug[NTOK][KAUG] (main 4096 columns) ------------
__global__ __launch_bounds__(256) void x_to_bf16(
    const float* __restrict__ x, unsigned short* __restrict__ Xb) {
  size_t i4 = (size_t)blockIdx.x * blockDim.x + threadIdx.x;
  const size_t total4 = (size_t)NTOK * IN_DIM / 4;
  if (i4 >= total4) return;
  int n  = (int)(i4 / (IN_DIM / 4));
  int k4 = (int)(i4 % (IN_DIM / 4));
  float4 f = reinterpret_cast<const float4*>(x)[i4];
  ushort4 o;
  o.x = f32_to_bf16_rte(f.x); o.y = f32_to_bf16_rte(f.y);
  o.z = f32_to_bf16_rte(f.z); o.w = f32_to_bf16_rte(f.w);
  *reinterpret_cast<ushort4*>(Xb + (size_t)n * KAUG + (size_t)k4 * 4) = o;
}

// --- prep 3: LoRA tail columns: Xaug[n, 4096+r] = (x @ A^T)[n, r] ----------
__global__ __launch_bounds__(256) void lora_xa_tail(
    const float* __restrict__ x, const float* __restrict__ lora_a,
    unsigned short* __restrict__ Xb) {
  int t = blockIdx.x * blockDim.x + threadIdx.x;
  if (t >= NTOK * KPAD) return;
  int n = t / KPAD, r = t % KPAD;
  float acc = 0.0f;
  if (r < RANK) {
    const float4* xr = reinterpret_cast<const float4*>(x + (size_t)n * IN_DIM);
    const float4* ar = reinterpret_cast<const float4*>(lora_a + (size_t)r * IN_DIM);
    for (int k = 0; k < IN_DIM / 4; ++k) {
      float4 a = xr[k], b = ar[k];
      acc += a.x * b.x + a.y * b.y + a.z * b.z + a.w * b.w;
    }
  }
  Xb[(size_t)n * KAUG + IN_DIM + r] = f32_to_bf16_rte(acc);
}

// --- async global -> LDS staging helpers (CDNA5 ASYNCcnt path) -------------
__device__ __forceinline__ void async_b128_to_lds(const unsigned short* g,
                                                  unsigned short* l) {
  // LDS flat addresses carry the LDS offset in addr[31:0] (aperture layout).
  unsigned lds_off = (unsigned)(unsigned long long)l;
  asm volatile("global_load_async_to_lds_b128 %0, %1, off"
               :: "v"(lds_off), "v"(g) : "memory");
}
__device__ __forceinline__ void wait_asynccnt0() {
  asm volatile("s_wait_asynccnt 0x0" ::: "memory");
}

// --- main GEMM: C[NTOK,OUT_DIM] = Xaug @ Waug^T, bf16 WMMA, f32 accum ------
// 256 threads = 8 wave32s (2 M x 4 N); each wave: 64x32 sub-tile = 4x2 grid
// of 16x16 accumulators. BK=64 stage = 2 k-substeps x 8 WMMA = 16 WMMA/stage.
__global__ __launch_bounds__(256) void qlora_gemm_bf16(
    const unsigned short* __restrict__ Xb, const unsigned short* __restrict__ Wb,
    float* __restrict__ out) {
  __shared__ __align__(16) unsigned short sX[2][BM * BK];  // 2 x 16 KB
  __shared__ __align__(16) unsigned short sW[2][BN * BK];  // 2 x 16 KB

  const int tid  = threadIdx.x;
  const int m0   = blockIdx.y * BM;
  const int n0   = blockIdx.x * BN;
  const int wave = tid >> 5;
  const int lane = tid & 31;
  const int wm   = (wave & 1) * 64;   // wave offset along M
  const int wn   = (wave >> 1) * 32;  // wave offset along N
  const int lrow = lane & 15;         // row within 16x16 fragment
  const int lhi  = lane >> 4;         // K-group (ISA 16-bit A/B lane split)

  v8f acc[4][2] = {};

  // issue async DMA for K-tile 0 into buffer 0
  // tile = 128 rows x 64 halfs = 1024 b128 chunks per operand
  #pragma unroll
  for (int c = tid; c < (BM * BK) / 8; c += 256) {
    int row = c >> 3, cc = c & 7;
    async_b128_to_lds(Xb + (size_t)(m0 + row) * KAUG + cc * 8,
                      &sX[0][row * BK + cc * 8]);
    async_b128_to_lds(Wb + (size_t)(n0 + row) * KAUG + cc * 8,
                      &sW[0][row * BK + cc * 8]);
  }

  int buf = 0;
  for (int kt = 0; kt < KTILES; ++kt) {
    wait_asynccnt0();     // our DMA into current buffer has landed in LDS
    __syncthreads();      // everyone's DMA has landed

    // L2 prefetch two stages ahead
    if (kt + 2 < KTILES) {
      const int kk2  = (kt + 2) * BK;
      const int prow = tid & 127;
      const unsigned short* pp = (tid < 128)
          ? Xb + (size_t)(m0 + prow) * KAUG + kk2
          : Wb + (size_t)(n0 + prow) * KAUG + kk2;
      __builtin_prefetch(pp, 0, 3);
    }

    // fire-and-forget DMA for next tile into the other buffer
    if (kt + 1 < KTILES) {
      const int kk = (kt + 1) * BK;
      #pragma unroll
      for (int c = tid; c < (BM * BK) / 8; c += 256) {
        int row = c >> 3, cc = c & 7;
        async_b128_to_lds(Xb + (size_t)(m0 + row) * KAUG + kk + cc * 8,
                          &sX[buf ^ 1][row * BK + cc * 8]);
        async_b128_to_lds(Wb + (size_t)(n0 + row) * KAUG + kk + cc * 8,
                          &sW[buf ^ 1][row * BK + cc * 8]);
      }
    }

    // compute: 2 k-substeps of 32; CDNA5 16-bit fragment layout:
    // lane<16 holds K {0..7,16..23}, lane>=16 holds K {8..15,24..31}
    union FragU { bf16x16 v; uint4 u[2]; };
    #pragma unroll
    for (int ks = 0; ks < 2; ++ks) {
      const int kb = ks * 32;
      bf16x16 afrag[4], bfrag[2];
      #pragma unroll
      for (int i = 0; i < 4; ++i) {
        const unsigned short* p =
            &sX[buf][(wm + i * 16 + lrow) * BK + kb + lhi * 8];
        FragU f;
        f.u[0] = *reinterpret_cast<const uint4*>(p);
        f.u[1] = *reinterpret_cast<const uint4*>(p + 16);
        afrag[i] = f.v;
      }
      #pragma unroll
      for (int j = 0; j < 2; ++j) {
        const unsigned short* p =
            &sW[buf][(wn + j * 16 + lrow) * BK + kb + lhi * 8];
        FragU f;
        f.u[0] = *reinterpret_cast<const uint4*>(p);
        f.u[1] = *reinterpret_cast<const uint4*>(p + 16);
        bfrag[j] = f.v;
      }
      #pragma unroll
      for (int i = 0; i < 4; ++i)
        #pragma unroll
        for (int j = 0; j < 2; ++j)
          acc[i][j] = __builtin_amdgcn_wmma_f32_16x16x32_bf16(
              false, afrag[i], false, bfrag[j], (short)0, acc[i][j],
              false, false);
    }

    buf ^= 1;
  }

  // C/D layout: lane L -> N = L%16; VGPR r -> M = r + 8*(L/16)
  #pragma unroll
  for (int i = 0; i < 4; ++i) {
    #pragma unroll
    for (int j = 0; j < 2; ++j) {
      const int n  = n0 + wn + j * 16 + lrow;
      const int mb = m0 + wm + i * 16 + lhi * 8;
      float* po = out + (size_t)mb * OUT_DIM + n;
      #pragma unroll
      for (int r = 0; r < 8; ++r)
        __builtin_nontemporal_store(acc[i][j][r], po + (size_t)r * OUT_DIM);
    }
  }
}

// ---------------------------------------------------------------------------
extern "C" void kernel_launch(void* const* d_in, const int* in_sizes, int n_in,
                              void* d_out, int out_size, void* d_ws, size_t ws_size,
                              hipStream_t stream) {
  const float* x       = (const float*)d_in[0];   // [8192, 4096] f32
  const int*   codes   = (const int*)d_in[1];     // [262144, 64] i32
  const float* scalers = (const float*)d_in[2];   // [262144] f32
  const float* lora_a  = (const float*)d_in[3];   // [16, 4096] f32
  const float* lora_b  = (const float*)d_in[4];   // [4096, 16] f32
  float* out = (float*)d_out;                     // [8192, 4096] f32

  // workspace: Waug bf16 (34.1 MB) then Xaug bf16 (68.2 MB) -> ~102.2 MB
  unsigned short* Wb = (unsigned short*)d_ws;
  unsigned short* Xb = Wb + (size_t)OUT_DIM * KAUG;

  {
    size_t total = (size_t)OUT_DIM * KAUG;
    nf4_dequant_w<<<dim3((unsigned)((total + 255) / 256)), 256, 0, stream>>>(
        codes, scalers, lora_b, Wb);
  }
  {
    size_t total4 = (size_t)NTOK * IN_DIM / 4;
    x_to_bf16<<<dim3((unsigned)((total4 + 255) / 256)), 256, 0, stream>>>(x, Xb);
  }
  lora_xa_tail<<<dim3(NTOK * KPAD / 256), 256, 0, stream>>>(x, lora_a, Xb);

  qlora_gemm_bf16<<<dim3(OUT_DIM / BN, NTOK / BM), 256, 0, stream>>>(Xb, Wb, out);
}
